// RWKV_83631603187895
// MI455X (gfx1250) — compile-verified, matrix operand order checked by reading, not run on hardware
//
#include <hip/hip_runtime.h>
#include <hip/hip_bf16.h>
#include <math.h>

// RWKV-7 style forward for MI455X (gfx1250, wave32).
// GEMMs: LDS-tiled v_wmma_f32_16x16x32_bf16 with GLOBAL_LOAD_ASYNC_TO_LDS
// staging (ASYNCcnt) for full tiles; fp32->bf16 at fragment build.
// Recurrent delta-rule scan: register-resident 64x64 state per (b,h).

#define BB 2
#define TT 1024
#define DDIM 1024
#define HH 16
#define KDIM 64
#define LL 2
#define VV 32000
#define SS (BB*TT)          // 2048 tokens

typedef __attribute__((ext_vector_type(16))) __bf16 v16bf;
typedef __attribute__((ext_vector_type(8)))  __bf16 v8bf;
typedef __attribute__((ext_vector_type(8)))  float  v8f;
typedef __attribute__((ext_vector_type(4)))  float  v4f;

__device__ __forceinline__ __bf16 cvt_bf16(float f) { return (__bf16)f; }

// ---------------------------------------------------------------------------
// LDS-tiled GEMM: C[M,N] = A[M,Kd] x B[Kd,N], row-major fp32 in/out.
// block = 256 threads (8 waves). C tile = 128(M) x 64(N).
// Wave w: waveM = w&3 (32 rows), waveN = w>>2 (32 cols) -> 2x2 accumulators.
// K-step = 32. A LDS [128x32] fp32; B LDS transposed [64 x 36] fp32.
// Full interior tiles staged with global_load_async_to_lds (ASYNCcnt);
// edge tiles (K/N tails) staged with guarded scalar copies (zero fill).
// Epilogue: C restaged through LDS for coalesced b128 stores.
// Requires M % 128 == 0 (true here: M = 2048 always), N % 16 == 0.
// ---------------------------------------------------------------------------
#define AF 32               // A lds row stride (floats)
#define BF 36               // B lds row stride (floats, padded)
#define SMEMF 8192          // floats: >= 128*AF + 64*BF (6400) and >= 128*64 (8192)

__global__ void __launch_bounds__(256) rwkv_gemm(
    const float* __restrict__ A, const float* __restrict__ Bm,
    float* __restrict__ C, int M, int Kd, int N) {
  __shared__ __align__(16) float smem[SMEMF];
  float* Al = smem;                // [128][AF]
  float* Bl = smem + 128 * AF;     // [64][BF]

  const int tid  = threadIdx.x;          // 0..255
  const int lane = tid & 31;
  const int wave = tid >> 5;             // 0..7
  const int wm   = (wave & 3) * 32;      // wave row offset in tile
  const int wn   = (wave >> 2) * 32;     // wave col offset in tile
  const int l15  = lane & 15;
  const int kb   = (lane >> 4) * 8;      // 0 or 8

  const int rowBase = blockIdx.y * 128;
  const int n0      = blockIdx.x * 64;

  v8f acc[2][2] = {};

  for (int k0 = 0; k0 < Kd; k0 += 32) {
    const bool full = (k0 + 32 <= Kd) && (n0 + 64 <= N);   // uniform per block
    if (full) {
      // ---- async staging: A 128x32 fp32 as coalesced b128 copies
#pragma unroll
      for (int e = 0; e < 4; ++e) {
        int idx = e * 256 + tid;           // 0..1023 float4 chunks
        int r   = idx >> 3;
        int c4  = (idx & 7) * 4;
        const float* gp = A + (size_t)(rowBase + r) * Kd + k0 + c4;
        unsigned lo = (unsigned)(unsigned long long)&Al[r * AF + c4];
        asm volatile("global_load_async_to_lds_b128 %0, %1, off"
                     :: "v"(lo), "v"(gp) : "memory");
      }
      // ---- async staging: B transposed (scatter) as b32 copies
#pragma unroll
      for (int e = 0; e < 8; ++e) {
        int idx = e * 256 + tid;           // 0..2047
        int kk  = idx >> 6;
        int nn  = idx & 63;
        const float* gp = Bm + (size_t)(k0 + kk) * N + n0 + nn;
        unsigned lo = (unsigned)(unsigned long long)&Bl[nn * BF + kk];
        asm volatile("global_load_async_to_lds_b32 %0, %1, off"
                     :: "v"(lo), "v"(gp) : "memory");
      }
      asm volatile("s_wait_asynccnt 0" ::: "memory");
    } else {
      // ---- guarded scalar staging with zero fill (no EXEC branches)
#pragma unroll
      for (int e = 0; e < 16; ++e) {
        int idx = e * 256 + tid;
        int r   = idx >> 5;
        int kk  = idx & 31;
        int kg  = k0 + kk;
        int ks  = (kg < Kd) ? kg : 0;
        float v = A[(size_t)(rowBase + r) * Kd + ks];
        Al[r * AF + kk] = (kg < Kd) ? v : 0.0f;
      }
#pragma unroll
      for (int e = 0; e < 8; ++e) {
        int idx = e * 256 + tid;
        int kk  = idx >> 6;
        int nn  = idx & 63;
        int kg  = k0 + kk;
        int cg  = n0 + nn;
        bool ok = (kg < Kd) && (cg < N);
        int ks  = (kg < Kd) ? kg : 0;
        int cs  = (cg < N)  ? cg : 0;
        float v = Bm[(size_t)ks * N + cs];
        Bl[nn * BF + kk] = ok ? v : 0.0f;
      }
    }
    __syncthreads();

    // ---- fragments from LDS (aligned b128 fp32 loads, cvt to bf16)
    v16bf afr[2], bfr[2];
#pragma unroll
    for (int mi = 0; mi < 2; ++mi) {
      const float* rp = &Al[(wm + mi * 16 + l15) * AF];
      v4f a0 = *reinterpret_cast<const v4f*>(rp + kb);
      v4f a1 = *reinterpret_cast<const v4f*>(rp + kb + 4);
      v4f a2 = *reinterpret_cast<const v4f*>(rp + 16 + kb);
      v4f a3 = *reinterpret_cast<const v4f*>(rp + 16 + kb + 4);
#pragma unroll
      for (int i = 0; i < 4; ++i) {
        afr[mi][i]      = cvt_bf16(a0[i]);
        afr[mi][4 + i]  = cvt_bf16(a1[i]);
        afr[mi][8 + i]  = cvt_bf16(a2[i]);
        afr[mi][12 + i] = cvt_bf16(a3[i]);
      }
    }
#pragma unroll
    for (int ni = 0; ni < 2; ++ni) {
      const float* rp = &Bl[(wn + ni * 16 + l15) * BF];
      v4f b0 = *reinterpret_cast<const v4f*>(rp + kb);
      v4f b1 = *reinterpret_cast<const v4f*>(rp + kb + 4);
      v4f b2 = *reinterpret_cast<const v4f*>(rp + 16 + kb);
      v4f b3 = *reinterpret_cast<const v4f*>(rp + 16 + kb + 4);
#pragma unroll
      for (int i = 0; i < 4; ++i) {
        bfr[ni][i]      = cvt_bf16(b0[i]);
        bfr[ni][4 + i]  = cvt_bf16(b1[i]);
        bfr[ni][8 + i]  = cvt_bf16(b2[i]);
        bfr[ni][12 + i] = cvt_bf16(b3[i]);
      }
    }
#pragma unroll
    for (int mi = 0; mi < 2; ++mi)
#pragma unroll
      for (int ni = 0; ni < 2; ++ni)
        acc[mi][ni] = __builtin_amdgcn_wmma_f32_16x16x32_bf16(
            false, afr[mi], false, bfr[ni], (short)0, acc[mi][ni], false, false);
    __syncthreads();
  }

  // ---- epilogue: restage C tile in LDS, then coalesced b128 stores
  float* Cs = smem;                       // reuse staging LDS: 128x64 fp32
#pragma unroll
  for (int mi = 0; mi < 2; ++mi) {
    int r0 = wm + mi * 16 + (lane >> 4) * 8;
#pragma unroll
    for (int ni = 0; ni < 2; ++ni) {
      int c = wn + ni * 16 + l15;
#pragma unroll
      for (int v = 0; v < 8; ++v)
        Cs[(r0 + v) * 64 + c] = acc[mi][ni][v];
    }
  }
  __syncthreads();
#pragma unroll
  for (int e = 0; e < 8; ++e) {
    int idx = e * 256 + tid;              // 0..2047 float4 chunks
    int r   = idx >> 4;                   // 0..127
    int c4  = (idx & 15) * 4;             // 0..60
    int col = n0 + c4;
    if (col < N) {                        // uniform per 16-col group (N%16==0)
      v4f val = *reinterpret_cast<const v4f*>(&Cs[r * 64 + c4]);
      *reinterpret_cast<v4f*>(&C[(size_t)(rowBase + r) * N + col]) = val;
    }
  }
}

// ---------------------------------------------------------------------------
// LayerNorm over D=1024: one block (256 thr) per token. Safe for in==out.
// ---------------------------------------------------------------------------
__global__ void rwkv_ln(const float* __restrict__ in, const float* __restrict__ sc,
                        const float* __restrict__ bi, float* __restrict__ outp) {
  const int rowBase = blockIdx.x * DDIM;
  const int tid = threadIdx.x;       // 256 threads, 4 elems each
  float v0[4];
  float s = 0.0f;
#pragma unroll
  for (int q = 0; q < 4; ++q) { v0[q] = in[rowBase + tid * 4 + q]; s += v0[q]; }
  __shared__ float red[256];
  red[tid] = s; __syncthreads();
  for (int st = 128; st > 0; st >>= 1) { if (tid < st) red[tid] += red[tid + st]; __syncthreads(); }
  float m = red[0] / (float)DDIM; __syncthreads();
  float vs = 0.0f;
#pragma unroll
  for (int q = 0; q < 4; ++q) { float d = v0[q] - m; vs += d * d; }
  red[tid] = vs; __syncthreads();
  for (int st = 128; st > 0; st >>= 1) { if (tid < st) red[tid] += red[tid + st]; __syncthreads(); }
  float inv = rsqrtf(red[0] / (float)DDIM + 1e-5f);
#pragma unroll
  for (int q = 0; q < 4; ++q) {
    int c = tid * 4 + q;
    outp[rowBase + c] = (v0[q] - m) * inv * sc[c] + bi[c];
  }
}

// ---------------------------------------------------------------------------
// Elementwise kernels (grid = ceil(n/256))
// ---------------------------------------------------------------------------
__global__ void rwkv_gather(const int* __restrict__ tok, const float* __restrict__ emb,
                            float* __restrict__ x, int n) {
  int idx = blockIdx.x * 256 + threadIdx.x;
  if (idx >= n) return;
  int s = idx / DDIM, c = idx % DDIM;
  x[idx] = emb[(size_t)tok[s] * DDIM + c];
}

__global__ void rwkv_shift(const float* __restrict__ xln, float* __restrict__ xp, int n) {
  int idx = blockIdx.x * 256 + threadIdx.x;
  if (idx >= n) return;
  int s = idx / DDIM;
  int t = s % TT;
  xp[idx] = (t == 0) ? 0.0f : xln[idx - DDIM];
}

__global__ void rwkv_lerpmu(const float* __restrict__ x, const float* __restrict__ xp,
                            const float* __restrict__ mu, float* __restrict__ outp, int n) {
  int idx = blockIdx.x * 256 + threadIdx.x;
  if (idx >= n) return;
  int c = idx % DDIM;
  float a = x[idx];
  outp[idx] = a + (xp[idx] - a) * mu[c];
}

__global__ void rwkv_ddfinal(const float* __restrict__ x, const float* __restrict__ xp,
                             const float* __restrict__ mu, const float* __restrict__ amt,
                             float* __restrict__ outp, int n) {
  int idx = blockIdx.x * 256 + threadIdx.x;
  if (idx >= n) return;
  int c = idx % DDIM;
  float a = x[idx];
  outp[idx] = a + (xp[idx] - a) * (mu[c] + amt[idx]);
}

__global__ void rwkv_tanh_ip(float* __restrict__ b, int n) {
  int idx = blockIdx.x * 256 + threadIdx.x;
  if (idx >= n) return;
  b[idx] = tanhf(b[idx]);
}

__global__ void rwkv_decay(const float* __restrict__ amt, const float* __restrict__ bias,
                           float* __restrict__ dec, float* __restrict__ ldec, int n) {
  int idx = blockIdx.x * 256 + threadIdx.x;
  if (idx >= n) return;
  int c = idx % DDIM;
  float z = -(bias[c] + amt[idx]);
  float sp = (z > 20.0f) ? z : log1pf(expf(z));      // softplus(-lld)
  float lld = -0.5f - sp;
  float ld = -expf(lld);                             // log(decay) in (-e^-0.5, 0)
  ldec[idx] = ld;
  dec[idx] = expf(ld);
}

__global__ void rwkv_addv(const float* __restrict__ a, const float* __restrict__ b,
                          float* __restrict__ o, int n) {
  int idx = blockIdx.x * 256 + threadIdx.x;
  if (idx >= n) return;
  o[idx] = a[idx] + b[idx];
}

__global__ void rwkv_sigb(const float* __restrict__ amt, const float* __restrict__ bias,
                          float* __restrict__ o, int n) {
  int idx = blockIdx.x * 256 + threadIdx.x;
  if (idx >= n) return;
  int c = idx % DDIM;
  o[idx] = 1.0f / (1.0f + expf(-(bias[c] + amt[idx])));
}

__global__ void rwkv_kupd(float* __restrict__ k, const float* __restrict__ iclr,
                          const float* __restrict__ mix, const float* __restrict__ ldec,
                          const float* __restrict__ omd, int n) {
  int idx = blockIdx.x * 256 + threadIdx.x;
  if (idx >= n) return;
  float kk = k[idx];
  kk = kk + mix[idx] * (kk * iclr[idx] - kk);
  kk *= expf(fminf(ldec[idx] * omd[idx], 0.0f));
  k[idx] = kk;
}

__global__ void rwkv_cminx(const float* __restrict__ x, const float* __restrict__ xp,
                           const float* __restrict__ lv, float* __restrict__ o, int n) {
  int idx = blockIdx.x * 256 + threadIdx.x;
  if (idx >= n) return;
  float a = x[idx];
  o[idx] = a + lv[idx] * (xp[idx] - a);
}

__global__ void rwkv_relusq(float* __restrict__ h, int n) {
  int idx = blockIdx.x * 256 + threadIdx.x;
  if (idx >= n) return;
  float t = fmaxf(h[idx], 0.0f);
  h[idx] = t * t;
}

// ---------------------------------------------------------------------------
// Per-head L2 normalize of deformed key. grid = S*H, block = 64.
// ---------------------------------------------------------------------------
__global__ void rwkv_dknorm(float* __restrict__ dk) {
  const size_t base = (size_t)blockIdx.x * KDIM;
  const int i = threadIdx.x;
  float v = dk[base + i];
  __shared__ float red[KDIM];
  red[i] = v * v; __syncthreads();
  for (int st = 32; st > 0; st >>= 1) { if (i < st) red[i] += red[i + st]; __syncthreads(); }
  float nrm = fmaxf(sqrtf(red[0]), 1e-12f);
  dk[base + i] = v / nrm;
}

// ---------------------------------------------------------------------------
// Delta-rule recurrence. grid = B*H, block = 64. Thread i owns state row i.
// state = state*diag(d) - (state@dk) (x) (iclr*dk) + v (x) k ; out = state@r
// ---------------------------------------------------------------------------
__global__ void rwkv_scan(const float* __restrict__ r, const float* __restrict__ k,
                          const float* __restrict__ v, const float* __restrict__ d,
                          const float* __restrict__ ic, const float* __restrict__ dk,
                          float* __restrict__ outp) {
  const int bh = blockIdx.x;
  const int b = bh / HH, h = bh % HH;
  const int i = threadIdx.x;
  __shared__ float sr[KDIM], sk[KDIM], sv[KDIM], sd[KDIM], si[KDIM], sdk[KDIM];
  float st[KDIM];
#pragma unroll
  for (int j = 0; j < KDIM; ++j) st[j] = 0.0f;

  for (int t = 0; t < TT; ++t) {
    size_t base = ((size_t)(b * TT + t)) * DDIM + h * KDIM + i;
    sr[i] = r[base]; sk[i] = k[base]; sv[i] = v[base];
    sd[i] = d[base]; si[i] = ic[base]; sdk[i] = dk[base];
    __syncthreads();
    float rem = 0.0f;
#pragma unroll
    for (int j = 0; j < KDIM; ++j) { st[j] *= sd[j]; rem += st[j] * sdk[j]; }
    float vi = sv[i];
    float o = 0.0f;
#pragma unroll
    for (int j = 0; j < KDIM; ++j) {
      st[j] += vi * sk[j] - rem * si[j] * sdk[j];
      o += st[j] * sr[j];
    }
    outp[base] = o;
    __syncthreads();
  }
}

// ---------------------------------------------------------------------------
// GroupNorm(H groups, eps=6.4e-4) + bonus + gate. grid = S*H, block = 64.
// ---------------------------------------------------------------------------
__global__ void rwkv_gn(const float* __restrict__ so, const float* __restrict__ r,
                        const float* __restrict__ k, const float* __restrict__ v,
                        const float* __restrict__ u, const float* __restrict__ gns,
                        const float* __restrict__ gnb, const float* __restrict__ gate,
                        float* __restrict__ outp) {
  const size_t base = (size_t)blockIdx.x * KDIM;
  const int h = blockIdx.x % HH;
  const int i = threadIdx.x;
  const int c = h * KDIM + i;
  float x = so[base + i];
  __shared__ float red[KDIM];
  red[i] = x; __syncthreads();
  for (int s = 32; s > 0; s >>= 1) { if (i < s) red[i] += red[i + s]; __syncthreads(); }
  float m = red[0] / (float)KDIM; __syncthreads();
  float dd = x - m;
  red[i] = dd * dd; __syncthreads();
  for (int s = 32; s > 0; s >>= 1) { if (i < s) red[i] += red[i + s]; __syncthreads(); }
  float var = red[0] / (float)KDIM; __syncthreads();
  red[i] = r[base + i] * k[base + i] * u[c]; __syncthreads();
  for (int s = 32; s > 0; s >>= 1) { if (i < s) red[i] += red[i + s]; __syncthreads(); }
  float bsum = red[0];
  float yn = dd * rsqrtf(var + 0.00064f);
  float y = yn * gns[c] + gnb[c] + bsum * v[base + i];
  outp[base + i] = y * gate[base + i];
}

// ---------------------------------------------------------------------------
// Host orchestration
// ---------------------------------------------------------------------------
static inline dim3 ewg(int n) { return dim3((n + 255) / 256); }

extern "C" void kernel_launch(void* const* d_in, const int* in_sizes, int n_in,
                              void* d_out, int out_size, void* d_ws, size_t ws_size,
                              hipStream_t stream) {
  (void)in_sizes; (void)n_in; (void)out_size; (void)ws_size;

  const int*   tokens = (const int*)d_in[0];
  const float* embed  = (const float*)d_in[1];
  const float* emb_s  = (const float*)d_in[2];
  const float* emb_b  = (const float*)d_in[3];
  const float* out_s  = (const float*)d_in[4];
  const float* out_b  = (const float*)d_in[5];
  const float* unemb  = (const float*)d_in[6];
  const float* tm_s   = (const float*)d_in[7];
  const float* tm_b   = (const float*)d_in[8];
  const float* dd_mu  = (const float*)d_in[9];
  const float* dd_w1  = (const float*)d_in[10];
  const float* dd_w2  = (const float*)d_in[11];
  const float* Wr     = (const float*)d_in[12];
  const float* Wk     = (const float*)d_in[13];
  const float* Wv     = (const float*)d_in[14];
  const float* Wo     = (const float*)d_in[15];
  const float* gw1    = (const float*)d_in[16];
  const float* gw2    = (const float*)d_in[17];
  const float* decb   = (const float*)d_in[18];
  const float* decw1  = (const float*)d_in[19];
  const float* decw2  = (const float*)d_in[20];
  const float* dkw1   = (const float*)d_in[21];
  const float* dkw2   = (const float*)d_in[22];
  const float* iclrb_p= (const float*)d_in[23];
  const float* iclrw1 = (const float*)d_in[24];
  const float* iclrw2 = (const float*)d_in[25];
  const float* im_b_p = (const float*)d_in[26];
  const float* imw1   = (const float*)d_in[27];
  const float* imw2   = (const float*)d_in[28];
  const float* od_b_p = (const float*)d_in[29];
  const float* odw1   = (const float*)d_in[30];
  const float* odw2   = (const float*)d_in[31];
  const float* u_p    = (const float*)d_in[32];
  const float* gns_p  = (const float*)d_in[33];
  const float* gnb_p  = (const float*)d_in[34];
  const float* cm_s   = (const float*)d_in[35];
  const float* cm_b   = (const float*)d_in[36];
  const float* W_in   = (const float*)d_in[37];
  const float* W_out  = (const float*)d_in[38];
  const float* W_lerp = (const float*)d_in[39];

  float* ws = (float*)d_ws;
  const size_t SD = (size_t)SS * DDIM;
  float* x      = ws +  0 * SD;
  float* xln    = ws +  1 * SD;
  float* xp     = ws +  2 * SD;
  float* xa     = ws +  3 * SD;
  float* xr     = ws +  4 * SD;
  float* xk     = ws +  5 * SD;
  float* xv     = ws +  6 * SD;
  float* xd     = ws +  7 * SD;
  float* rb     = ws +  8 * SD;
  float* kb     = ws +  9 * SD;
  float* vb     = ws + 10 * SD;
  float* gateb  = ws + 11 * SD;
  float* tsm    = ws + 12 * SD;   // small LoRA hidden (<= S*128)
  float* amt    = ws + 13 * SD;
  float* decayb = ws + 14 * SD;
  float* ldec   = ws + 15 * SD;
  float* dkb    = ws + 16 * SD;
  float* iclrb  = ws + 17 * SD;
  float* mixb   = ws + 18 * SD;
  float* omdb   = ws + 19 * SD;
  float* so     = ws + 20 * SD;
  float* gno    = ws + 21 * SD;
  float* proj   = ws + 22 * SD;
  float* hbuf   = ws + 23 * SD;   // S x 4D (spans 4 slots)
  float* inx    = ws + 27 * SD;
  float* lerpv  = ws + 28 * SD;

  const int nSD = SS * DDIM;

  auto GEMM = [&](const float* A, const float* Bm, float* C, int M, int Kd, int N) {
    dim3 blk(256, 1, 1);
    dim3 grd((N + 63) / 64, M / 128, 1);   // M == 2048 always (mult of 128)
    rwkv_gemm<<<grd, blk, 0, stream>>>(A, Bm, C, M, Kd, N);
  };
  auto LORA = [&](const float* xin, const float* w1, const float* w2, int R, float* dst) {
    GEMM(xin, w1, tsm, SS, DDIM, R);
    rwkv_tanh_ip<<<ewg(SS * R), 256, 0, stream>>>(tsm, SS * R);
    GEMM(tsm, w2, dst, SS, R, DDIM);
  };

  // ---- embedding + LN ----
  rwkv_gather<<<ewg(nSD), 256, 0, stream>>>(tokens, embed, x, nSD);
  rwkv_ln<<<SS, 256, 0, stream>>>(x, emb_s, emb_b, x);

  for (int l = 0; l < LL; ++l) {
    const size_t D2 = (size_t)DDIM * DDIM;
    const float* Wr_l  = Wr + l * D2;
    const float* Wk_l  = Wk + l * D2;
    const float* Wv_l  = Wv + l * D2;
    const float* Wo_l  = Wo + l * D2;

    // ===== time mixer =====
    rwkv_ln<<<SS, 256, 0, stream>>>(x, tm_s + (size_t)l * DDIM, tm_b + (size_t)l * DDIM, xln);
    rwkv_shift<<<ewg(nSD), 256, 0, stream>>>(xln, xp, nSD);

    float* dsts[4] = { xr, xk, xv, xd };
    for (int i = 0; i < 4; ++i) {
      const float* mu  = dd_mu + ((size_t)l * 4 + i) * DDIM;
      const float* w1  = dd_w1 + ((size_t)l * 4 + i) * DDIM * 32;
      const float* w2  = dd_w2 + ((size_t)l * 4 + i) * 32 * DDIM;
      rwkv_lerpmu<<<ewg(nSD), 256, 0, stream>>>(xln, xp, mu, xa, nSD);
      GEMM(xa, w1, tsm, SS, DDIM, 32);
      rwkv_tanh_ip<<<ewg(SS * 32), 256, 0, stream>>>(tsm, SS * 32);
      GEMM(tsm, w2, amt, SS, 32, DDIM);
      rwkv_ddfinal<<<ewg(nSD), 256, 0, stream>>>(xln, xp, mu, amt, dsts[i], nSD);
    }

    GEMM(xr, Wr_l, rb, SS, DDIM, DDIM);
    GEMM(xk, Wk_l, kb, SS, DDIM, DDIM);
    GEMM(xv, Wv_l, vb, SS, DDIM, DDIM);

    // gate = tanh(xr @ gw1) @ gw2
    LORA(xr, gw1 + (size_t)l * DDIM * 128, gw2 + (size_t)l * 128 * DDIM, 128, gateb);

    // decay
    LORA(xd, decw1 + (size_t)l * DDIM * 64, decw2 + (size_t)l * 64 * DDIM, 64, amt);
    rwkv_decay<<<ewg(nSD), 256, 0, stream>>>(amt, decb + (size_t)l * DDIM, decayb, ldec, nSD);

    // deformed key (uses pre-update k)
    LORA(xk, dkw1 + (size_t)l * DDIM * 16, dkw2 + (size_t)l * 16 * DDIM, 16, amt);
    rwkv_addv<<<ewg(nSD), 256, 0, stream>>>(kb, amt, dkb, nSD);
    rwkv_dknorm<<<SS * HH, KDIM, 0, stream>>>(dkb);

    // iclr / iclr_mix (from xd) and omd (from xk)
    LORA(xd, iclrw1 + (size_t)l * DDIM * 16, iclrw2 + (size_t)l * 16 * DDIM, 16, amt);
    rwkv_sigb<<<ewg(nSD), 256, 0, stream>>>(amt, iclrb_p + (size_t)l * DDIM, iclrb, nSD);
    LORA(xd, imw1 + (size_t)l * DDIM * 16, imw2 + (size_t)l * 16 * DDIM, 16, amt);
    rwkv_sigb<<<ewg(nSD), 256, 0, stream>>>(amt, im_b_p + (size_t)l * DDIM, mixb, nSD);
    LORA(xk, odw1 + (size_t)l * DDIM * 16, odw2 + (size_t)l * 16 * DDIM, 16, amt);
    rwkv_sigb<<<ewg(nSD), 256, 0, stream>>>(amt, od_b_p + (size_t)l * DDIM, omdb, nSD);

    // k <- lerp(k, k*iclr, mix) * exp(min(log_decay*omd, 0))
    rwkv_kupd<<<ewg(nSD), 256, 0, stream>>>(kb, iclrb, mixb, ldec, omdb, nSD);

    // recurrence
    rwkv_scan<<<BB * HH, KDIM, 0, stream>>>(rb, kb, vb, decayb, iclrb, dkb, so);

    // groupnorm + bonus + gate, then output projection and residual
    rwkv_gn<<<SS * HH, KDIM, 0, stream>>>(so, rb, kb, vb, u_p + (size_t)l * HH * KDIM,
                                          gns_p + (size_t)l * DDIM, gnb_p + (size_t)l * DDIM,
                                          gateb, gno);
    GEMM(gno, Wo_l, proj, SS, DDIM, DDIM);
    rwkv_addv<<<ewg(nSD), 256, 0, stream>>>(xln, proj, x, nSD);

    // ===== channel mixer =====
    rwkv_ln<<<SS, 256, 0, stream>>>(x, cm_s + (size_t)l * DDIM, cm_b + (size_t)l * DDIM, xln);
    rwkv_shift<<<ewg(nSD), 256, 0, stream>>>(xln, xp, nSD);
    GEMM(xln, W_lerp + l * D2, lerpv, SS, DDIM, DDIM);
    rwkv_cminx<<<ewg(nSD), 256, 0, stream>>>(xln, xp, lerpv, inx, nSD);
    GEMM(inx, W_in + (size_t)l * DDIM * 4 * DDIM, hbuf, SS, DDIM, 4 * DDIM);
    rwkv_relusq<<<ewg(SS * 4 * DDIM), 256, 0, stream>>>(hbuf, SS * 4 * DDIM);
    GEMM(hbuf, W_out + (size_t)l * 4 * DDIM * DDIM, proj, SS, 4 * DDIM, DDIM);
    rwkv_addv<<<ewg(nSD), 256, 0, stream>>>(xln, proj, x, nSD);
  }

  // ---- final LN + unembed ----
  rwkv_ln<<<SS, 256, 0, stream>>>(x, out_s, out_b, xln);
  GEMM(xln, unemb, (float*)d_out, SS, DDIM, VV);
}
